// Transformer_14370960573065
// MI455X (gfx1250) — matrix operand order repaired and checked by reading, not compiled, not run on hardware
//
#include <hip/hip_runtime.h>

// ---------------- problem constants (match reference) ----------------
static constexpr int BB = 32, T = 128, D = 384, H = 6, L = 6, V = 32000, HS = 64;
static constexpr int M  = BB * T;          // 4096 rows in the token stream
static constexpr int D4 = 4 * D;           // 1536

typedef __bf16 bf16;
typedef bf16  v16bf __attribute__((ext_vector_type(16)));
typedef bf16  v8bf  __attribute__((ext_vector_type(8)));
typedef float v8f   __attribute__((ext_vector_type(8)));
typedef unsigned int u32x4 __attribute__((ext_vector_type(4)));
typedef int          i32x8 __attribute__((ext_vector_type(8)));
typedef int          i32x4 __attribute__((ext_vector_type(4)));

#define DEV __device__ __forceinline__

DEV bf16 f2bf(float f) {
  unsigned u = __builtin_bit_cast(unsigned, f);
  u += 0x7FFFu + ((u >> 16) & 1u);               // round-to-nearest-even
  unsigned short h = (unsigned short)(u >> 16);
  return __builtin_bit_cast(bf16, h);
}

// ---- WMMA fragment loaders (contiguous-k 16-bit order, wave32) ----
// A 16x32 (MxK): lanes 0-15 hold row M=lane, halves0-15 = K0-15
//                lanes 16-31 hold row M=lane-16, halves0-15 = K16-31
DEV v16bf load_a_frag(const bf16* A, int lda, int row0, int k0, int lane) {
  int r  = row0 + (lane & 15);
  int kb = k0 + ((lane & 16) ? 16 : 0);
  const bf16* p = A + (size_t)r * lda + kb;
  v8bf lo = *(const v8bf*)p;
  v8bf hi = *(const v8bf*)(p + 8);
  return __builtin_shufflevector(lo, hi, 0,1,2,3,4,5,6,7,8,9,10,11,12,13,14,15);
}
// B 32x16 (KxN) from N-major storage BT[n*ldb + k]:
// lanes 0-15: column N=lane, K=k0..k0+15 ; lanes 16-31: column N=lane-16, K=k0+16..k0+31
DEV v16bf load_b_frag(const bf16* BT, int ldb, int n0, int k0, int lane) {
  int c  = n0 + (lane & 15);
  int kb = k0 + ((lane & 16) ? 16 : 0);
  return *(const v16bf*)(BT + (size_t)c * ldb + kb);
}
DEV v8f wmma_bf16(v16bf a, v16bf b, v8f c) {
  return __builtin_amdgcn_wmma_f32_16x16x32_bf16(false, a, false, b, (short)0, c,
                                                 false, false);
}

// ---- Tensor Data Mover: DMA a dense [rows x rowlen] bf16 tile into LDS ----
// D# per CDNA5 ISA ch.8: group0 = count/lds_addr/global_addr/type,
// group1 = data_size + tensor/tile dims + dim0 stride. 2-D tile -> groups 2/3 zero.
// This toolchain exposes the 6-arg builtin (g0, g1, g2, g3, g4, cpol).
DEV void tdm_load_tile(const bf16* gsrc, unsigned lds_addr,
                       int rows, int rowlen, int row_stride_elems) {
  unsigned long long ga = (unsigned long long)(uintptr_t)gsrc;
  u32x4 g0;
  g0[0] = 1u;                                            // count=1, no gather
  g0[1] = lds_addr;                                      // LDS byte address
  g0[2] = (unsigned)(ga & 0xFFFFFFFFu);                  // global_addr[31:0]
  g0[3] = (unsigned)((ga >> 32) & 0x01FFFFFFu)           // global_addr[56:32]
        | (2u << 30);                                    // type=2 ("image")
  unsigned long long rl = (unsigned)rowlen, rw = (unsigned)rows;
  unsigned long long st = (unsigned long long)(unsigned)row_stride_elems;
  unsigned long long q0 = (1ull << 16)                   // data_size=1 -> 2 bytes
                        | ((rl & 0xFFFFull) << 48);      // tensor_dim0[15:0]
  unsigned long long q1 = (rl >> 16)                     // tensor_dim0[31:16]
                        | (rw << 16)                     // tensor_dim1
                        | ((rl & 0xFFFFull) << 48);      // tile_dim0
  unsigned long long q2 = (rw & 0xFFFFull)               // tile_dim1
                        | ((st & 0xFFFFFFFFull) << 32);  // tensor_dim0_stride[31:0]
  unsigned long long q3 = (st >> 32) & 0xFFFFull;        // tensor_dim0_stride[47:32]
  i32x8 g1;
  g1[0] = (int)q0; g1[1] = (int)(q0 >> 32);
  g1[2] = (int)q1; g1[3] = (int)(q1 >> 32);
  g1[4] = (int)q2; g1[5] = (int)(q2 >> 32);
  g1[6] = (int)q3; g1[7] = (int)(q3 >> 32);
  i32x4 gz4 = {0, 0, 0, 0};
  i32x8 gz8 = {0, 0, 0, 0, 0, 0, 0, 0};
  __builtin_amdgcn_tensor_load_to_lds(g0, g1, gz4, gz4, gz8, 0);
}

// ---------------- weight convert + transpose: dst[n*K+k] = bf16(src[k*N+n]) ----
__global__ void __launch_bounds__(256)
convert_tr_kernel(const float* __restrict__ src, bf16* __restrict__ dst,
                  int K, int N, size_t src_stride, size_t dst_stride) {
  const float* s = src + (size_t)blockIdx.y * src_stride;
  bf16*        d = dst + (size_t)blockIdx.y * dst_stride;
  int total = K * N;
  for (int i = blockIdx.x * 256 + threadIdx.x; i < total; i += gridDim.x * 256) {
    int n = i / K, k = i - n * K;
    d[(size_t)n * K + k] = f2bf(s[(size_t)k * N + n]);
  }
}

// ---------------- embedding: x[bt,d] = tok_emb[tok[bt],d] + pos_emb[t,d] ----
__global__ void __launch_bounds__(128)
embed_kernel(const int* __restrict__ tokens, const float* __restrict__ tok_emb,
             const float* __restrict__ pos_emb, float* __restrict__ x) {
  int bt = blockIdx.x;
  int tok = tokens[bt];
  int t = bt & (T - 1);
  for (int d = threadIdx.x; d < D; d += 128)
    x[(size_t)bt * D + d] = tok_emb[(size_t)tok * D + d] + pos_emb[(size_t)t * D + d];
}

// ---------------- LayerNorm: f32 in -> bf16 out ----
__global__ void __launch_bounds__(128)
ln_kernel(const float* __restrict__ x, const float* __restrict__ g,
          const float* __restrict__ be, bf16* __restrict__ out) {
  __shared__ float red[128];
  int row = blockIdx.x, tid = threadIdx.x;
  const float* xr = x + (size_t)row * D;
  float v0 = xr[tid], v1 = xr[tid + 128], v2 = xr[tid + 256];
  red[tid] = v0 + v1 + v2;
  __syncthreads();
  for (int o = 64; o > 0; o >>= 1) { if (tid < o) red[tid] += red[tid + o]; __syncthreads(); }
  float mean = red[0] * (1.f / D);
  __syncthreads();
  float d0 = v0 - mean, d1 = v1 - mean, d2 = v2 - mean;
  red[tid] = d0 * d0 + d1 * d1 + d2 * d2;
  __syncthreads();
  for (int o = 64; o > 0; o >>= 1) { if (tid < o) red[tid] += red[tid + o]; __syncthreads(); }
  float inv = rsqrtf(red[0] * (1.f / D) + 1e-5f);
  bf16* orow = out + (size_t)row * D;
  orow[tid]       = f2bf(d0 * inv * g[tid]       + be[tid]);
  orow[tid + 128] = f2bf(d1 * inv * g[tid + 128] + be[tid + 128]);
  orow[tid + 256] = f2bf(d2 * inv * g[tid + 256] + be[tid + 256]);
}

// ---------------- generic WMMA GEMM: C[M,N] = A[M,K] @ B[K,N] (B stored N-major) ----
// Block tile 128x64 (8 waves, each 16 rows x 64 cols = 4 WMMA tiles).
// B k-slab (64x32 bf16) staged via the Tensor Data Mover into LDS, double-buffered:
// wave 0 issues tensor_load_to_lds for slab k+1 while all waves run WMMA on slab k.
// FLAGS: 1=bias, 2=relu, 4=residual add (f32), 8=bf16 output (else f32)
template <int F>
__global__ void __launch_bounds__(256)
gemm_kernel(const bf16* __restrict__ A, const bf16* __restrict__ BT,
            const float* __restrict__ bias, const float* __restrict__ resid,
            float* __restrict__ Cf, bf16* __restrict__ Cb,
            int Mm, int N, int K) {
  __shared__ bf16 lb[2][64][32];                   // double-buffered B tile, 8 KB
  const int lane = threadIdx.x & 31, wave = threadIdx.x >> 5;
  const int row0 = blockIdx.y * 128 + wave * 16;
  const int nb   = blockIdx.x * 64;
  const unsigned lds0 = (unsigned)(uintptr_t)&lb[0][0][0];
  const unsigned lds1 = (unsigned)(uintptr_t)&lb[1][0][0];
  v8f acc[4] = {};
  if (threadIdx.x == 0)                            // prime buffer 0 (TENSORcnt)
    tdm_load_tile(BT + (size_t)nb * K, lds0, 64, 32, K);
  int buf = 0;
  for (int k0 = 0; k0 < K; k0 += 32) {
    if (threadIdx.x == 0) __builtin_amdgcn_s_wait_tensorcnt(0);
    __syncthreads();                               // tile[buf] visible to all waves
    if (threadIdx.x == 0 && k0 + 32 < K)           // async prefetch of next slab
      tdm_load_tile(BT + (size_t)nb * K + k0 + 32, buf ? lds0 : lds1, 64, 32, K);
    v16bf a = load_a_frag(A, K, row0, k0, lane);
#pragma unroll
    for (int t = 0; t < 4; ++t)
      acc[t] = wmma_bf16(a, load_b_frag(&lb[buf][0][0], 32, t * 16, 0, lane), acc[t]);
    __syncthreads();                               // everyone done with tile[buf]
    buf ^= 1;
  }
  const int rbase = row0 + ((lane & 16) ? 8 : 0);
#pragma unroll
  for (int t = 0; t < 4; ++t) {
    int c = nb + t * 16 + (lane & 15);
    float bv = (F & 1) ? bias[c] : 0.f;
#pragma unroll
    for (int r = 0; r < 8; ++r) {
      size_t idx = (size_t)(rbase + r) * N + c;
      float v = acc[t][r] + bv;
      if (F & 2) v = v > 0.f ? v : 0.f;
      if (F & 4) v += resid[idx];
      if (F & 8) Cb[idx] = f2bf(v);
      else       Cf[idx] = v;
    }
  }
}

// ---------------- fused causal attention for one (b,h): 128x64 heads ----
__global__ void __launch_bounds__(128)
attn_kernel(const bf16* __restrict__ qb, const bf16* __restrict__ kmat,
            const bf16* __restrict__ vmat, bf16* __restrict__ ob) {
  __shared__ bf16 vt[HS][T];        // V^T: vt[e][t], 16 KB
  __shared__ bf16 pS[4][16][T];     // per-wave softmax(P) strip, 16 KB
  const int bh = blockIdx.x, b = bh / H, h = bh % H;
  const int lane = threadIdx.x & 31, wave = threadIdx.x >> 5;
  const size_t base = (size_t)b * T * D + (size_t)h * HS;
  const bf16* Q  = qb   + base;
  const bf16* Kp = kmat + base;   // row-major K is exactly the N-major B operand for Q·K^T
  const bf16* Vp = vmat + base;

  for (int i = threadIdx.x; i < HS * T; i += 128) {
    int e = i >> 7, t = i & (T - 1);
    vt[e][t] = Vp[(size_t)t * D + e];
  }
  __syncthreads();

  const int colit  = lane & 15;
  const int rowadd = (lane & 16) ? 8 : 0;

  for (int ss = wave; ss < T / 16; ss += 4) {       // two strips per wave, uniform trip count
    const int qrow0 = ss * 16;
    // ---- S = (Q K^T) * scale, 16x128 strip ----
    v8f s[8] = {};
#pragma unroll
    for (int k0 = 0; k0 < HS; k0 += 32) {
      v16bf a = load_a_frag(Q, D, qrow0, k0, lane);
#pragma unroll
      for (int t = 0; t < 8; ++t)
        s[t] = wmma_bf16(a, load_b_frag(Kp, D, t * 16, k0, lane), s[t]);
    }
    // ---- causal mask + softmax per row (rows live across a 16-lane group) ----
#pragma unroll
    for (int vv = 0; vv < 8; ++vv) {
      int row = qrow0 + vv + rowadd;
      float m = -1e30f;
#pragma unroll
      for (int t = 0; t < 8; ++t) {
        int col = t * 16 + colit;
        float val = s[t][vv] * 0.125f;               // HS^-0.5
        val = (col <= row) ? val : -1e30f;
        s[t][vv] = val;
        m = fmaxf(m, val);
      }
      for (int off = 1; off < 16; off <<= 1) m = fmaxf(m, __shfl_xor(m, off, 32));
      float sum = 0.f;
#pragma unroll
      for (int t = 0; t < 8; ++t) {
        float e = (t * 16 + colit <= row) ? __expf(s[t][vv] - m) : 0.f;
        s[t][vv] = e;
        sum += e;
      }
      for (int off = 1; off < 16; off <<= 1) sum += __shfl_xor(sum, off, 32);
      float inv = 1.f / sum;
#pragma unroll
      for (int t = 0; t < 8; ++t)
        pS[wave][vv + rowadd][t * 16 + colit] = f2bf(s[t][vv] * inv);
    }
    __syncthreads();
    // ---- O = P @ V  (16x64 strip) ----
    v8f o[4] = {};
#pragma unroll
    for (int k0 = 0; k0 < T; k0 += 32) {
      v16bf a = load_a_frag(&pS[wave][0][0], T, 0, k0, lane);
#pragma unroll
      for (int t = 0; t < 4; ++t)
        o[t] = wmma_bf16(a, load_b_frag(&vt[0][0], T, t * 16, k0, lane), o[t]);
    }
#pragma unroll
    for (int t = 0; t < 4; ++t) {
      int e = t * 16 + colit;
#pragma unroll
      for (int r = 0; r < 8; ++r) {
        int row = qrow0 + rowadd + r;
        ob[(size_t)(b * T + row) * D + h * HS + e] = f2bf(o[t][r]);
      }
    }
    __syncthreads();
  }
}

// ---------------- final row softmax over V: single-pass online max/sum ----
// 2 reads + 1 write of the 524 MB logits instead of 3 reads + 2 writes.
__global__ void __launch_bounds__(256)
softmaxv_kernel(float* __restrict__ out) {
  __shared__ float mm[256], ss[256];
  int tid = threadIdx.x;
  float* p = out + (size_t)blockIdx.x * V;
  float m = -1e30f, s = 0.f;
  for (int i = tid; i < V; i += 256) {
    float v = p[i];
    if (v > m) { s = s * __expf(m - v) + 1.f; m = v; }
    else       { s += __expf(v - m); }
  }
  mm[tid] = m; ss[tid] = s;
  __syncthreads();
  for (int o = 128; o > 0; o >>= 1) {
    if (tid < o) {
      float m2 = mm[tid + o], s2 = ss[tid + o];
      float mx = fmaxf(mm[tid], m2);
      ss[tid] = ss[tid] * __expf(mm[tid] - mx) + s2 * __expf(m2 - mx);
      mm[tid] = mx;
    }
    __syncthreads();
  }
  m = mm[0];
  float inv = 1.f / ss[0];
  for (int i = tid; i < V; i += 256) p[i] = __expf(p[i] - m) * inv;
}

// ---------------- host launcher ----------------
static inline int cdiv(int a, int b) { return (a + b - 1) / b; }

extern "C" void kernel_launch(void* const* d_in, const int* in_sizes, int n_in,
                              void* d_out, int out_size, void* d_ws, size_t ws_size,
                              hipStream_t stream) {
  (void)in_sizes; (void)n_in; (void)out_size; (void)ws_size;
  const int*   tokens  = (const int*)  d_in[0];
  const float* tok_emb = (const float*)d_in[1];
  const float* pos_emb = (const float*)d_in[2];
  const float* Wq  = (const float*)d_in[3];
  const float* Wk  = (const float*)d_in[4];
  const float* Wv  = (const float*)d_in[5];
  const float* Wo  = (const float*)d_in[6];
  const float* bo  = (const float*)d_in[7];
  const float* W1  = (const float*)d_in[8];
  const float* b1  = (const float*)d_in[9];
  const float* W2  = (const float*)d_in[10];
  const float* b2  = (const float*)d_in[11];
  const float* g1  = (const float*)d_in[12];
  const float* be1 = (const float*)d_in[13];
  const float* g2  = (const float*)d_in[14];
  const float* be2 = (const float*)d_in[15];
  const float* gf  = (const float*)d_in[16];
  const float* bef = (const float*)d_in[17];
  const float* Wout= (const float*)d_in[18];
  const float* bout= (const float*)d_in[19];
  float* out = (float*)d_out;

  // workspace carve-up (256B aligned slabs)
  char* wsp = (char*)d_ws;
  auto alloc = [&](size_t bytes) -> char* {
    char* p = wsp; wsp += (bytes + 255) & ~(size_t)255; return p;
  };
  bf16* WqT   = (bf16*)alloc((size_t)L * D * D * 2);        // [L][H*HS][D]
  bf16* WkT   = (bf16*)alloc((size_t)L * D * D * 2);
  bf16* WvT   = (bf16*)alloc((size_t)L * D * D * 2);
  bf16* WoT   = (bf16*)alloc((size_t)L * D * D * 2);        // [L][N=D][K=D]
  bf16* W1T   = (bf16*)alloc((size_t)L * D * D4 * 2);       // [L][N=4D][K=D]
  bf16* W2T   = (bf16*)alloc((size_t)L * D4 * D * 2);       // [L][N=D][K=4D]
  bf16* WoutT = (bf16*)alloc((size_t)V * D * 2);            // [N=V][K=D]
  float* x    = (float*)alloc((size_t)M * D * 4);           // residual stream f32
  bf16* hb    = (bf16*)alloc((size_t)M * D * 2);            // LN output
  bf16* qB    = (bf16*)alloc((size_t)M * D * 2);
  bf16* kB    = (bf16*)alloc((size_t)M * D * 2);
  bf16* vB    = (bf16*)alloc((size_t)M * D * 2);
  bf16* oB    = (bf16*)alloc((size_t)M * D * 2);
  bf16* uB    = (bf16*)alloc((size_t)M * D4 * 2);
  bf16* xfB   = (bf16*)alloc((size_t)M * D * 2);

  // ---- one-time weight conversion (f32 -> bf16, transposed to N-major) ----
  convert_tr_kernel<<<dim3(cdiv(D * HS, 256), L * H), 256, 0, stream>>>(Wq, WqT, D, HS, (size_t)D * HS, (size_t)D * HS);
  convert_tr_kernel<<<dim3(cdiv(D * HS, 256), L * H), 256, 0, stream>>>(Wk, WkT, D, HS, (size_t)D * HS, (size_t)D * HS);
  convert_tr_kernel<<<dim3(cdiv(D * HS, 256), L * H), 256, 0, stream>>>(Wv, WvT, D, HS, (size_t)D * HS, (size_t)D * HS);
  convert_tr_kernel<<<dim3(cdiv(D * D,  256), L),     256, 0, stream>>>(Wo, WoT, D, D,  (size_t)D * D,  (size_t)D * D);
  convert_tr_kernel<<<dim3(cdiv(D * D4, 256), L),     256, 0, stream>>>(W1, W1T, D, D4, (size_t)D * D4, (size_t)D * D4);
  convert_tr_kernel<<<dim3(cdiv(D4 * D, 256), L),     256, 0, stream>>>(W2, W2T, D4, D, (size_t)D4 * D, (size_t)D4 * D);
  convert_tr_kernel<<<dim3(cdiv(D * V,  256), 1),     256, 0, stream>>>(Wout, WoutT, D, V, 0, 0);

  // ---- embedding ----
  embed_kernel<<<M, 128, 0, stream>>>(tokens, tok_emb, pos_emb, x);

  const dim3 gD(D / 64, M / 128);      // N=384 GEMMs
  const dim3 gD4(D4 / 64, M / 128);    // N=1536 GEMM
  const dim3 gV(V / 64, M / 128);      // logits GEMM

  for (int l = 0; l < L; ++l) {
    const bf16* wq = WqT + (size_t)l * D * D;
    const bf16* wk = WkT + (size_t)l * D * D;
    const bf16* wv = WvT + (size_t)l * D * D;
    const bf16* wo = WoT + (size_t)l * D * D;
    const bf16* w1 = W1T + (size_t)l * D * D4;
    const bf16* w2 = W2T + (size_t)l * D4 * D;

    ln_kernel<<<M, 128, 0, stream>>>(x, g1 + l * D, be1 + l * D, hb);
    gemm_kernel<8><<<gD, 256, 0, stream>>>(hb, wq, nullptr, nullptr, nullptr, qB, M, D, D);
    gemm_kernel<8><<<gD, 256, 0, stream>>>(hb, wk, nullptr, nullptr, nullptr, kB, M, D, D);
    gemm_kernel<8><<<gD, 256, 0, stream>>>(hb, wv, nullptr, nullptr, nullptr, vB, M, D, D);
    attn_kernel<<<BB * H, 128, 0, stream>>>(qB, kB, vB, oB);
    gemm_kernel<5><<<gD, 256, 0, stream>>>(oB, wo, bo + l * D, x, x, nullptr, M, D, D);
    ln_kernel<<<M, 128, 0, stream>>>(x, g2 + l * D, be2 + l * D, hb);
    gemm_kernel<11><<<gD4, 256, 0, stream>>>(hb, w1, b1 + l * D4, nullptr, nullptr, uB, M, D4, D);
    gemm_kernel<5><<<gD, 256, 0, stream>>>(uB, w2, b2 + l * D, x, x, nullptr, M, D, D4);
  }

  ln_kernel<<<M, 128, 0, stream>>>(x, gf, bef, xfB);
  gemm_kernel<1><<<gV, 256, 0, stream>>>(xfB, WoutT, bout, nullptr, out, nullptr, M, V, D);
  softmaxv_kernel<<<M, 256, 0, stream>>>(out);
}
